// DependencyParser_72593537237715
// MI455X (gfx1250) — matrix operand order repaired and checked by reading, not compiled, not run on hardware
//
#include <hip/hip_runtime.h>
#include <hip/hip_bf16.h>
#include <math.h>
#include <stdint.h>

typedef _Float16 f16;
typedef _Float16 v16h __attribute__((ext_vector_type(16)));
typedef _Float16 v8h  __attribute__((ext_vector_type(8)));
typedef float    v8f  __attribute__((ext_vector_type(8)));

#define HID  512
#define SEQ  512
#define EMB  345
#define EMBP 352      // EMB padded to multiple of 32
#define G4   2048     // 4*HID
#define NN   513      // SEQ+1
#define NP   528      // NN padded to 33*16

__device__ __forceinline__ v16h mk16(v8h lo, v8h hi) {
  v16h r;
#pragma unroll
  for (int e = 0; e < 8; ++e) { r[e] = lo[e]; r[e + 8] = hi[e]; }
  return r;
}
__device__ __forceinline__ float sigm(float x) { return 1.f / (1.f + expf(-x)); }

// async global->LDS copy of one 16B chunk (CDNA5 ASYNCcnt path)
__device__ __forceinline__ void async_b128(unsigned lds_off, unsigned long gaddr) {
  asm volatile("global_load_async_to_lds_b128 %0, %1, off"
               :: "v"(lds_off), "v"(gaddr) : "memory");
}
__device__ __forceinline__ void wait_async0() {
  asm volatile("s_wait_asynccnt 0" ::: "memory");
}

// ---------------- conversion / init kernels ----------------
__global__ void pack_f16_k(f16* __restrict__ dst, const float* __restrict__ src,
                           int rows, int ksrc, int kp, int ldsrc) {
  long idx = (long)blockIdx.x * blockDim.x + threadIdx.x;
  long total = (long)rows * kp;
  if (idx >= total) return;
  int r = (int)(idx / kp);
  int k = (int)(idx - (long)r * kp);
  dst[idx] = (k < ksrc) ? (f16)src[(long)r * ldsrc + k] : (f16)0.f;
}

__global__ void zero_f16_k(f16* __restrict__ dst, long n) {
  long i = (long)blockIdx.x * blockDim.x + threadIdx.x;
  if (i < n) dst[i] = (f16)0.f;
}

// ---------------- generic WMMA GEMM: C = A(f16,MxK) @ B(f16,NxK)^T + bias ----------------
__global__ __launch_bounds__(32) void gemm_wmma_k(
    float* __restrict__ C, const f16* __restrict__ A, const f16* __restrict__ B,
    const float* __restrict__ bias, int Kp, int ldc) {
  const int lane  = threadIdx.x & 31;
  const int l15   = lane & 15;
  const int khalf = lane >> 4;
  const int mbase = blockIdx.y * 16;
  const int nbase = blockIdx.x * 16;
  v8f acc = {};
  const f16* ap = A + (long)(mbase + l15) * Kp + khalf * 8;   // A-frag: K runs at +0 and +16
  const f16* bp = B + (long)(nbase + l15) * Kp + khalf * 8;   // B-frag: K runs at +0 and +16
  for (int kc = 0; kc < Kp; kc += 32) {
    v16h a = mk16(*(const v8h*)(ap + kc), *(const v8h*)(ap + kc + 16));
    v16h b = mk16(*(const v8h*)(bp + kc), *(const v8h*)(bp + kc + 16));
    acc = __builtin_amdgcn_wmma_f32_16x16x32_f16(false, a, false, b, (short)0, acc, false, false);
  }
  const int n = nbase + l15;
  const float bv = bias ? bias[n] : 0.f;
  const int m0 = mbase + khalf * 8;            // C: M = r + 8*(lane/16), N = lane%16
#pragma unroll
  for (int r = 0; r < 8; ++r)
    C[(long)(m0 + r) * ldc + n] = acc[r] + bv;
}

// ---------------- LSTM recurrence: blockIdx.x = direction, 512 threads = 16 waves ----------------
__global__ __launch_bounds__(512) void lstm_wmma_k(
    const float* __restrict__ zin /*2 x SEQ x 4H*/,
    const f16*  __restrict__ whh /*2 x 4H x H (f16)*/,
    float* __restrict__ out /*SEQ x 2H*/) {
  __shared__ f16   h16[HID];
  __shared__ float zacc[G4];
  const int dir  = blockIdx.x;
  const int tid  = threadIdx.x;
  const int lane = tid & 31, wave = tid >> 5;
  const int l15  = lane & 15, khalf = lane >> 4;
  const float* zd = zin + (long)dir * SEQ * G4;
  const f16*   W  = whh + (long)dir * G4 * HID;
  float c = 0.f;
  h16[tid] = (f16)0.f;
  __syncthreads();

  for (int s = 0; s < SEQ; ++s) {
    const int t = dir ? (SEQ - 1 - s) : s;
    // z = Whh @ h : 128 gate tiles of 16, 8 per wave; only B column 0 is live
#pragma unroll 1
    for (int g8 = 0; g8 < 8; ++g8) {
      const int gt = wave * 8 + g8;
      const f16* ap = W + (long)(gt * 16 + l15) * HID + khalf * 8;
      v8f acc = {};
      const v8h zz = {};
      for (int kc = 0; kc < HID; kc += 32) {
        v16h a = mk16(*(const v8h*)(ap + kc), *(const v8h*)(ap + kc + 16));
        v16h b;
        if (l15 == 0) {
          const f16* hp = &h16[kc + khalf * 8];
          b = mk16(*(const v8h*)hp, *(const v8h*)(hp + 16));
        } else {
          b = mk16(zz, zz);
        }
        acc = __builtin_amdgcn_wmma_f32_16x16x32_f16(false, a, false, b, (short)0, acc, false, false);
      }
      if (l15 == 0) {
#pragma unroll
        for (int r = 0; r < 8; ++r) zacc[gt * 16 + khalf * 8 + r] = acc[r];
      }
    }
    __syncthreads();
    // gates: thread tid owns hidden unit tid (gate order i,f,g,o)
    const float* zt = zd + (long)t * G4;
    const float zi = zacc[tid]           + zt[tid];
    const float zf = zacc[HID + tid]     + zt[HID + tid];
    const float zg = zacc[2 * HID + tid] + zt[2 * HID + tid];
    const float zo = zacc[3 * HID + tid] + zt[3 * HID + tid];
    c = sigm(zf) * c + sigm(zi) * tanhf(zg);
    const float h = sigm(zo) * tanhf(c);
    out[(long)t * (2 * HID) + dir * HID + tid] = h;
    h16[tid] = (f16)h;
    __syncthreads();
  }
}

// ---------------- fused pair MLP: h1=relu(A[i]+B[j]+b1) -> h2 -> h3 -> S ----------------
__global__ __launch_bounds__(256) void pair_mlp_k(
    const float* __restrict__ Ab, const float* __restrict__ Bb,
    const float* __restrict__ b1, const f16* __restrict__ W2g,
    const float* __restrict__ b2, const f16* __restrict__ W3g,
    const float* __restrict__ b3, const float* __restrict__ W4,
    const float* __restrict__ b4, float* __restrict__ S) {
  __shared__ f16   W2s[128 * 256];      // 64 KB, async-staged
  __shared__ f16   W3s[64 * 128];       // 16 KB, async-staged
  __shared__ f16   h2s[8 * 16 * 128];   // 32 KB (per-wave staging)
  __shared__ float b1s[256], b2s[128], b3s[64], W4s[64];
  const int tid = threadIdx.x, lane = tid & 31, wave = tid >> 5;
  const int l15 = lane & 15, khalf = lane >> 4;
  const int ib = blockIdx.y * 16, jb = blockIdx.x * 16;

  // stage W2 (4096 x 16B) and W3 (1024 x 16B) via async global->LDS
  {
    const unsigned l2 = (unsigned)(uintptr_t)(&W2s[0]);
    const unsigned l3 = (unsigned)(uintptr_t)(&W3s[0]);
    const unsigned long g2 = (unsigned long)W2g;
    const unsigned long g3 = (unsigned long)W3g;
#pragma unroll
    for (int it = 0; it < 16; ++it) {
      const unsigned off = (unsigned)(tid + 256 * it) * 16u;
      async_b128(l2 + off, g2 + off);
    }
#pragma unroll
    for (int it = 0; it < 4; ++it) {
      const unsigned off = (unsigned)(tid + 256 * it) * 16u;
      async_b128(l3 + off, g3 + off);
    }
  }
  b1s[tid] = b1[tid];
  if (tid < 128) b2s[tid] = b2[tid];
  if (tid < 64) { b3s[tid] = b3[tid]; W4s[tid] = W4[tid]; }
  wait_async0();
  __syncthreads();
  const float bias4 = b4[0];

  for (int rep = 0; rep < 2; ++rep) {
    const int tile = rep * 8 + wave;        // i-offset within block
    const int i = ib + tile;
    const int j = jb + l15;                 // pair row (M) = j-offset
    // build h1 directly in WMMA A-fragment layout (16 pairs x K=256)
    v16h frag[8];
    const float* Ar = Ab + (long)i * 256;
    const float* Br = Bb + (long)j * 256;
#pragma unroll
    for (int kc = 0; kc < 8; ++kc) {
      const int kb = kc * 32 + khalf * 8;
#pragma unroll
      for (int e = 0; e < 16; ++e) {
        const int k = kb + (e & 7) + ((e >> 3) << 4);
        frag[kc][e] = (f16)fmaxf(Ar[k] + Br[k] + b1s[k], 0.f);
      }
    }
    // h2 = relu(h1 @ W2^T + b2), stage as f16 in LDS
    f16* h2w = &h2s[wave * 2048];
#pragma unroll 1
    for (int n0 = 0; n0 < 128; n0 += 16) {
      v8f acc = {};
      const f16* bp = &W2s[(n0 + l15) * 256 + khalf * 8];
#pragma unroll
      for (int kc = 0; kc < 8; ++kc) {
        v16h b = mk16(*(const v8h*)(bp + kc * 32), *(const v8h*)(bp + kc * 32 + 16));
        acc = __builtin_amdgcn_wmma_f32_16x16x32_f16(false, frag[kc], false, b, (short)0, acc, false, false);
      }
      const int n = n0 + l15;
#pragma unroll
      for (int r = 0; r < 8; ++r)
        h2w[(khalf * 8 + r) * 128 + n] = (f16)fmaxf(acc[r] + b2s[n], 0.f);
    }
    // h3 = relu(h2 @ W3^T + b3), fused with S = h3 . W4 + b4
    float sp[8] = {0.f, 0.f, 0.f, 0.f, 0.f, 0.f, 0.f, 0.f};
#pragma unroll 1
    for (int n0 = 0; n0 < 64; n0 += 16) {
      v8f acc = {};
      const f16* ap = h2w + l15 * 128 + khalf * 8;
      const f16* bp = &W3s[(n0 + l15) * 128 + khalf * 8];
#pragma unroll
      for (int kc = 0; kc < 4; ++kc) {
        v16h a = mk16(*(const v8h*)(ap + kc * 32), *(const v8h*)(ap + kc * 32 + 16));
        v16h b = mk16(*(const v8h*)(bp + kc * 32), *(const v8h*)(bp + kc * 32 + 16));
        acc = __builtin_amdgcn_wmma_f32_16x16x32_f16(false, a, false, b, (short)0, acc, false, false);
      }
      const int n = n0 + l15;
      const float w4 = W4s[n], bb = b3s[n];
#pragma unroll
      for (int r = 0; r < 8; ++r) sp[r] += fmaxf(acc[r] + bb, 0.f) * w4;
    }
    // butterfly-reduce S partials over the 16-lane half-groups
#pragma unroll
    for (int r = 0; r < 8; ++r) {
#pragma unroll
      for (int off = 1; off < 16; off <<= 1) sp[r] += __shfl_xor(sp[r], off, 32);
    }
    if (l15 == 0) {
#pragma unroll
      for (int r = 0; r < 8; ++r)
        S[(long)i * NP + jb + khalf * 8 + r] = sp[r] + bias4;
    }
  }
}

// ---------------- softmax per row + per-row loss partial ----------------
__global__ __launch_bounds__(256) void softmax_loss_k(
    const float* __restrict__ S, const int* __restrict__ heads,
    float* __restrict__ outp, float* __restrict__ lossp) {
  __shared__ float red[256];
  const int i = blockIdx.x, tid = threadIdx.x;
  const float* row = S + (long)i * NP;
  float mx = -1e30f;
  for (int j = tid; j < NN; j += 256) mx = fmaxf(mx, row[j]);
  red[tid] = mx; __syncthreads();
  for (int k = 128; k > 0; k >>= 1) { if (tid < k) red[tid] = fmaxf(red[tid], red[tid + k]); __syncthreads(); }
  mx = red[0]; __syncthreads();
  const int hd = heads[i];
  float sum = 0.f, lp = 0.f;
  for (int j = tid; j < NN; j += 256) {
    const float s = row[j];
    sum += expf(s - mx);
    const float d = s - ((j == hd) ? 100.f : 0.f);
    lp += d * d;
  }
  red[tid] = sum; __syncthreads();
  for (int k = 128; k > 0; k >>= 1) { if (tid < k) red[tid] += red[tid + k]; __syncthreads(); }
  sum = red[0]; __syncthreads();
  red[tid] = lp; __syncthreads();
  for (int k = 128; k > 0; k >>= 1) { if (tid < k) red[tid] += red[tid + k]; __syncthreads(); }
  if (tid == 0) lossp[i] = red[0];
  const float inv = 1.f / sum;
  for (int j = tid; j < NN; j += 256) outp[1 + (long)i * NN + j] = expf(row[j] - mx) * inv;
}

__global__ __launch_bounds__(256) void loss_final_k(const float* __restrict__ lossp,
                                                    float* __restrict__ outp) {
  __shared__ float red[256];
  const int tid = threadIdx.x;
  float s = 0.f;
  for (int i = tid; i < NN; i += 256) s += lossp[i];
  red[tid] = s; __syncthreads();
  for (int k = 128; k > 0; k >>= 1) { if (tid < k) red[tid] += red[tid + k]; __syncthreads(); }
  if (tid == 0) outp[0] = red[0] / ((float)NN * (float)NN);
}

// ---------------- host orchestration ----------------
extern "C" void kernel_launch(void* const* d_in, const int* in_sizes, int n_in,
                              void* d_out, int out_size, void* d_ws, size_t ws_size,
                              hipStream_t stream) {
  (void)in_sizes; (void)n_in; (void)out_size; (void)ws_size;
  const float* x    = (const float*)d_in[0];
  const int*   heads= (const int*)  d_in[1];
  const float* Wih0 = (const float*)d_in[2];
  const float* Whh0 = (const float*)d_in[3];
  const float* b0   = (const float*)d_in[4];
  const float* Wihr = (const float*)d_in[5];
  const float* Whhr = (const float*)d_in[6];
  const float* b_r  = (const float*)d_in[7];
  const float* W1   = (const float*)d_in[8];
  const float* b1   = (const float*)d_in[9];
  const float* W2   = (const float*)d_in[10];
  const float* b2   = (const float*)d_in[11];
  const float* W3   = (const float*)d_in[12];
  const float* b3   = (const float*)d_in[13];
  const float* W4   = (const float*)d_in[14];
  const float* b4   = (const float*)d_in[15];
  float* out = (float*)d_out;

  char* p = (char*)d_ws;
  auto take = [&](size_t bytes) { char* r = p; p += (bytes + 255) & ~size_t(255); return r; };
  f16* x16     = (f16*)take(512L * EMBP * 2);
  f16* wih0_16 = (f16*)take(2L * G4 * EMBP * 2);
  f16* whh0_16 = (f16*)take(2L * G4 * HID * 2);
  f16* wihr16  = (f16*)take(6L * G4 * 1024 * 2);
  f16* whhr16  = (f16*)take(6L * G4 * HID * 2);
  f16* w1a16   = (f16*)take(256L * 1024 * 2);
  f16* w1b16   = (f16*)take(256L * 1024 * 2);
  f16* w2_16   = (f16*)take(128L * 256 * 2);
  f16* w3_16   = (f16*)take(64L * 128 * 2);
  f16* in16    = (f16*)take(512L * 1024 * 2);
  f16* av16    = (f16*)take((long)NP * 1024 * 2);
  float* zin   = (float*)take(2L * 512 * G4 * 4);
  float* outA  = (float*)take(512L * 1024 * 4);
  float* outB  = (float*)take(512L * 1024 * 4);
  float* Abuf  = (float*)take((long)NP * 256 * 4);
  float* Bbuf  = (float*)take((long)NP * 256 * 4);
  float* scr   = (float*)take((long)NP * NP * 4);
  float* lossp = (float*)take((long)NN * 4);

  auto packs = [&](f16* dst, const float* src, int rows, int ks, int kp, int ld) {
    long tot = (long)rows * kp;
    pack_f16_k<<<dim3((unsigned)((tot + 255) / 256)), dim3(256), 0, stream>>>(dst, src, rows, ks, kp, ld);
  };
  packs(x16, x, 512, EMB, EMBP, EMB);
  packs(wih0_16, Wih0, 2 * G4, EMB, EMBP, EMB);
  packs(whh0_16, Whh0, 2 * G4, HID, HID, HID);
  packs(wihr16, Wihr, 6 * G4, 1024, 1024, 1024);
  packs(whhr16, Whhr, 6 * G4, HID, HID, HID);
  packs(w1a16, W1, 256, 1024, 1024, 2048);
  packs(w1b16, W1 + 1024, 256, 1024, 1024, 2048);
  packs(w2_16, W2, 128, 256, 256, 256);
  packs(w3_16, W3, 64, 128, 128, 128);

  // layer 0: zin = x @ Wih^T + b, then recurrence
  for (int d = 0; d < 2; ++d)
    gemm_wmma_k<<<dim3(G4 / 16, 512 / 16), 32, 0, stream>>>(
        zin + (long)d * 512 * G4, x16, wih0_16 + (long)d * G4 * EMBP,
        b0 + (long)d * G4, EMBP, G4);
  lstm_wmma_k<<<2, 512, 0, stream>>>(zin, whh0_16, outA);

  float* prev = outA; float* cur = outB;
  for (int l = 0; l < 3; ++l) {
    packs(in16, prev, 512, 1024, 1024, 1024);
    for (int d = 0; d < 2; ++d)
      gemm_wmma_k<<<dim3(G4 / 16, 512 / 16), 32, 0, stream>>>(
          zin + (long)d * 512 * G4, in16, wihr16 + (long)(l * 2 + d) * G4 * 1024,
          b_r + (long)(l * 2 + d) * G4, 1024, G4);
    lstm_wmma_k<<<2, 512, 0, stream>>>(zin, whhr16 + (long)l * 2 * G4 * HID, cur);
    float* t = prev; prev = cur; cur = t;
  }

  // avail = [0 ; lstm_out ; zero-pad to 528] in f16
  { long tot = (long)NP * 1024;
    zero_f16_k<<<dim3((unsigned)((tot + 255) / 256)), 256, 0, stream>>>(av16, tot); }
  packs(av16 + 1024, prev, 512, 1024, 1024, 1024);

  // biaffine precompute: A = avail @ W1[:, :1024]^T ; B = avail @ W1[:, 1024:]^T
  gemm_wmma_k<<<dim3(256 / 16, NP / 16), 32, 0, stream>>>(Abuf, av16, w1a16, nullptr, 1024, 256);
  gemm_wmma_k<<<dim3(256 / 16, NP / 16), 32, 0, stream>>>(Bbuf, av16, w1b16, nullptr, 1024, 256);

  // fused pair MLP -> scores
  pair_mlp_k<<<dim3(NP / 16, NP / 16), 256, 0, stream>>>(
      Abuf, Bbuf, b1, w2_16, b2, w3_16, b3, W4, b4, scr);

  // softmax rows + loss
  softmax_loss_k<<<NN, 256, 0, stream>>>(scr, heads, out, lossp);
  loss_final_k<<<1, 256, 0, stream>>>(lossp, out);
}
